// ProxyNCA_prob_mixup_70308614636137
// MI455X (gfx1250) — matrix operand order repaired and checked by reading, not compiled, not run on hardware
//
#include <hip/hip_runtime.h>

// ---- CDNA5 vector types -----------------------------------------------------
typedef __attribute__((ext_vector_type(16))) __bf16       v16bf;
typedef __attribute__((ext_vector_type(8)))  float        v8f;
typedef __attribute__((ext_vector_type(4)))  unsigned int u32x4;
typedef __attribute__((ext_vector_type(4)))  float        f32x4;

// ---- problem constants (match reference) -----------------------------------
#define NROWS     65536
#define NB_CLASS  512
#define SZ_EMBED  512
#define SHIFTS    4
#define TILE_M    16
#define WPB       8              // waves per block in the main kernel
#define LDS_STRIDE 260           // dwords per tile row: 512 halfs + 8 pad halfs
#define TILE_DW   (TILE_M * LDS_STRIDE)   // 4160 dwords = 16,640 B per tile

// ---- helpers ---------------------------------------------------------------
__device__ __forceinline__ unsigned f2bf(float f) {           // RNE fp32->bf16
    unsigned u = __float_as_uint(f);
    unsigned r = u + 0x7FFFu + ((u >> 16) & 1u);
    return r >> 16;
}
__device__ __forceinline__ float bf2f(unsigned h) {           // bf16->fp32
    return __uint_as_float(h << 16);
}
// Deterministic stand-in for jax.random.beta(key(42),1,1): Beta(1,1)=U(0,1).
__device__ __forceinline__ float lam_of(unsigned j) {
    unsigned long long z = (unsigned long long)j * 0x9E3779B97F4A7C15ull
                         + 0xD1B54A32D192ED03ull;
    z ^= z >> 33; z *= 0xFF51AFD7ED558CCDull; z ^= z >> 33;
    return (float)((unsigned)(z >> 40)) * (1.0f / 16777216.0f);
}

// Issue this thread's share (4 async b128 copies) of B-tile `tct` into the LDS
// buffer starting at dword `buf_dw`. 256 threads x 4 issues x 16B = 16 KB.
__device__ __forceinline__ void stage_B_async(const unsigned* __restrict__ Pb,
                                              int tct, int buf_dw, int tid) {
    const unsigned long long pb = (unsigned long long)Pb;
    #pragma unroll
    for (int i = 0; i < 4; ++i) {
        const unsigned chunk = (unsigned)(i * 256 + tid);   // 0..1023
        const unsigned row   = chunk >> 6;                  // class row in tile
        const unsigned sub   = chunk & 63u;                 // 16B chunk in row
        const unsigned voff  = (unsigned)(tct * 16 + (int)row) * 1024u + sub * 16u;
        const unsigned ldsoff = (unsigned)(buf_dw + (int)row * LDS_STRIDE) * 4u
                              + sub * 16u;
        asm volatile("global_load_async_to_lds_b128 %0, %1, %2"
                     :: "v"(ldsoff), "v"(voff), "s"(pb) : "memory");
    }
}

// ---- kernel 1: Pn = 3*normalize(proxies) -> bf16 row-major; zero accum -----
__global__ __launch_bounds__(256)
void k_prep_proxies(const float* __restrict__ P, unsigned* __restrict__ Pb,
                    float* __restrict__ accum) {
    __shared__ float red[256];
    const int c = blockIdx.x, t = threadIdx.x;
    const float v0 = P[c * SZ_EMBED + 2 * t];
    const float v1 = P[c * SZ_EMBED + 2 * t + 1];
    red[t] = v0 * v0 + v1 * v1;
    __syncthreads();
    for (int s = 128; s > 0; s >>= 1) {
        if (t < s) red[t] += red[t + s];
        __syncthreads();
    }
    const float scale = 3.0f / fmaxf(sqrtf(red[0]), 1e-12f);
    Pb[c * 256 + t] = f2bf(v0 * scale) | (f2bf(v1 * scale) << 16);
    if (c == 0 && t == 0) *accum = 0.0f;
}

// ---- kernel 2: Xn = 3*normalize(X) -> bf16, one wave per row ---------------
__global__ __launch_bounds__(256)
void k_prep_x(const float* __restrict__ X, unsigned* __restrict__ Xb) {
    const int wave = threadIdx.x >> 5, lane = threadIdx.x & 31;
    const int row  = blockIdx.x * 8 + wave;
    const f32x4* p = (const f32x4*)(X + row * SZ_EMBED + lane * 16);
    f32x4 a = p[0], b = p[1], c = p[2], d = p[3];
    float ss = a.x*a.x + a.y*a.y + a.z*a.z + a.w*a.w
             + b.x*b.x + b.y*b.y + b.z*b.z + b.w*b.w
             + c.x*c.x + c.y*c.y + c.z*c.z + c.w*c.w
             + d.x*d.x + d.y*d.y + d.z*d.z + d.w*d.w;
    #pragma unroll
    for (int m = 1; m < 32; m <<= 1) ss += __shfl_xor(ss, m, 32);
    const float s = 3.0f / fmaxf(sqrtf(ss), 1e-12f);
    float f[16] = {a.x,a.y,a.z,a.w, b.x,b.y,b.z,b.w,
                   c.x,c.y,c.z,c.w, d.x,d.y,d.z,d.w};
    unsigned o[8];
    #pragma unroll
    for (int q = 0; q < 8; ++q)
        o[q] = f2bf(f[2*q] * s) | (f2bf(f[2*q+1] * s) << 16);
    u32x4* out = (u32x4*)(Xb + row * 256 + lane * 8);
    out[0] = (u32x4){o[0], o[1], o[2], o[3]};
    out[1] = (u32x4){o[4], o[5], o[6], o[7]};
}

// ---- kernel 3: fused GEMM + row loss ---------------------------------------
// Per block (8 waves): each wave owns one 16-row tile of Xall. A is staged in
// LDS, lifted into 128 register-resident VGPRs, then buffers 0/1 of the SAME
// LDS array are reused as two 16.6KB B buffers. B column tiles (16 classes x
// 512 K, bf16) are staged ONCE per block (amortized over 128 A rows) with
// global_load_async_to_lds_b128, double-buffered and overlapped with the
// 16-WMMA compute of the previous tile. The hot loop has no global loads:
// 2x ds_load_b128 + 1x v_wmma_f32_16x16x32_bf16 per step.
__global__ __launch_bounds__(WPB * 32)
void k_main(const int* __restrict__ T, const unsigned* __restrict__ Xb,
            const unsigned* __restrict__ Pb, float* __restrict__ accum) {
    __shared__ unsigned lds[WPB * TILE_DW];   // 133,120 B (gfx1250: 320KB/WG)

    const int tid  = threadIdx.x;
    const int wave = tid >> 5;
    const int lane = tid & 31;
    const int hi   = lane >> 4;               // half-wave select
    const int nl   = lane & 15;
    unsigned* A = &lds[wave * TILE_DW];

    const int tile     = blockIdx.x * WPB + wave;
    const int row_base = tile * TILE_M;
    const bool virt    = row_base >= NROWS;

    // -------- stage the 16x512 bf16 A tile into LDS (padded rows) ----------
    if (!virt) {
        const unsigned* src = Xb + row_base * 256 + lane * 8;
        #pragma unroll 4
        for (int r = 0; r < TILE_M; ++r) {
            const u32x4* gp = (const u32x4*)(src + r * 256);
            u32x4 lo = gp[0], hh = gp[1];
            u32x4* dp = (u32x4*)&A[r * LDS_STRIDE + lane * 8];
            dp[0] = lo; dp[1] = hh;
        }
    } else {
        const int jb = row_base - NROWS;
        for (int r = 0; r < TILE_M; ++r) {
            const int j  = jb + r;
            const int j2 = (j + SHIFTS) & (NROWS - 1);
            const float lam = lam_of((unsigned)j), ml = 1.0f - lam;
            const u32x4* p1 = (const u32x4*)(Xb + j  * 256 + lane * 8);
            const u32x4* p2 = (const u32x4*)(Xb + j2 * 256 + lane * 8);
            u32x4 a0 = p1[0], a1 = p1[1], b0 = p2[0], b1 = p2[1];
            unsigned ua[8] = {a0.x,a0.y,a0.z,a0.w, a1.x,a1.y,a1.z,a1.w};
            unsigned ub[8] = {b0.x,b0.y,b0.z,b0.w, b1.x,b1.y,b1.z,b1.w};
            float mix[16]; float ss = 0.0f;
            #pragma unroll
            for (int q = 0; q < 8; ++q) {
                float m0 = lam * bf2f(ua[q] & 0xFFFFu) + ml * bf2f(ub[q] & 0xFFFFu);
                float m1 = lam * bf2f(ua[q] >> 16)     + ml * bf2f(ub[q] >> 16);
                mix[2*q] = m0; mix[2*q+1] = m1;
                ss += m0 * m0 + m1 * m1;
            }
            #pragma unroll
            for (int m = 1; m < 32; m <<= 1) ss += __shfl_xor(ss, m, 32);
            const float s = 3.0f / fmaxf(sqrtf(ss), 1e-12f);
            unsigned o[8];
            #pragma unroll
            for (int q = 0; q < 8; ++q)
                o[q] = f2bf(mix[2*q] * s) | (f2bf(mix[2*q+1] * s) << 16);
            u32x4* dp = (u32x4*)&A[r * LDS_STRIDE + lane * 8];
            dp[0] = (u32x4){o[0], o[1], o[2], o[3]};
            dp[1] = (u32x4){o[4], o[5], o[6], o[7]};
        }
    }
    __syncthreads();

    // -------- lift all 16 A fragments into registers ------------------------
    // A layout (16-bit A 16x32): lane holds M=nl, K = ks*32 + hi*8 (+0..7,+16..23)
    union frag { u32x4 q[2]; v16bf v; };
    frag av[16];
    #pragma unroll
    for (int ks = 0; ks < 16; ++ks) {
        const u32x4* ap = (const u32x4*)&A[nl * LDS_STRIDE + ks * 16 + hi * 4];
        av[ks].q[0] = ap[0];
        av[ks].q[1] = ap[2];
    }
    __syncthreads();     // all waves done reading A -> buffers 0/1 free for B

    // -------- per-lane soft-label metadata (C/D layout: VGPR r -> M=r+8*hi) -
    int t1v[8], t2v[8]; float w1v[8], w2v[8];
    #pragma unroll
    for (int r = 0; r < 8; ++r) {
        const int m = r + hi * 8;
        if (!virt) {
            t1v[r] = T[row_base + m]; w1v[r] = 1.0f;
            t2v[r] = -1;              w2v[r] = 0.0f;
        } else {
            const int j  = row_base - NROWS + m;
            const int j2 = (j + SHIFTS) & (NROWS - 1);
            const float lam = lam_of((unsigned)j);
            t1v[r] = T[j];  w1v[r] = lam;
            t2v[r] = T[j2]; w2v[r] = 1.0f - lam;
        }
    }

    // -------- async prologue: stage B0 -> buf0, B1 -> buf1 ------------------
    stage_B_async(Pb, 0, 0,       tid);
    stage_B_async(Pb, 1, TILE_DW, tid);
    asm volatile("s_wait_asynccnt 0x4" ::: "memory");   // B0's 4 done (in order)
    __syncthreads();

    // -------- 32 column tiles x 16 K-steps of bf16 WMMA ---------------------
    float rsum[8] = {0,0,0,0,0,0,0,0};
    float tacc[8] = {0,0,0,0,0,0,0,0};

    for (int ct = 0; ct < 32; ++ct) {
        const int c = ct * 16 + nl;                      // this lane's class
        const unsigned* B = &lds[(ct & 1) * TILE_DW];
        v8f acc = {};
        #pragma unroll
        for (int ks = 0; ks < 16; ++ks) {
            // B fragment: lane holds column N=nl (= staged row nl),
            // K = ks*32 + hi*8 (+0..7,+16..23)
            const u32x4* bp = (const u32x4*)&B[nl * LDS_STRIDE + ks * 16 + hi * 4];
            frag ub;
            ub.q[0] = bp[0];
            ub.q[1] = bp[2];
            acc = __builtin_amdgcn_wmma_f32_16x16x32_bf16(
                false, av[ks].v, false, ub.v, (short)0, acc, false, false);
        }
        #pragma unroll
        for (int r = 0; r < 8; ++r) {
            const float d = fmaxf(18.0f - 2.0f * acc[r], 0.0f); // |x|=|p|=3
            rsum[r] += __expf(-d);                 // -d in [-36,0]: no underflow
            tacc[r] += (c == t1v[r]) ? w1v[r] * d : 0.0f;
            tacc[r] += (c == t2v[r]) ? w2v[r] * d : 0.0f;
        }
        __syncthreads();                  // all waves done reading buf[ct&1]
        if (ct < 30) {
            stage_B_async(Pb, ct + 2, (ct & 1) * TILE_DW, tid);
            asm volatile("s_wait_asynccnt 0x4" ::: "memory"); // B[ct+1] done
        } else {
            asm volatile("s_wait_asynccnt 0x0" ::: "memory"); // drain tail
        }
        __syncthreads();                  // buf[(ct+1)&1] staged for everyone
    }

    // -------- reduce across the 16 lanes of each half-wave ------------------
    float part = 0.0f;
    #pragma unroll
    for (int r = 0; r < 8; ++r) {
        float s = rsum[r], t = tacc[r];
        #pragma unroll
        for (int m = 1; m < 16; m <<= 1) {
            s += __shfl_xor(s, m, 32);
            t += __shfl_xor(t, m, 32);
        }
        part += t + __logf(s);   // loss_row = <T,D> + log sum exp(-D)
    }
    part += __shfl_xor(part, 16, 32);            // combine the two halves
    if (lane == 0) atomicAdd(accum, part);
}

// ---- kernel 4: mean --------------------------------------------------------
__global__ void k_finalize(const float* __restrict__ accum, float* __restrict__ out) {
    out[0] = accum[0] * (1.0f / (2.0f * (float)NROWS));
}

// ---- host glue -------------------------------------------------------------
extern "C" void kernel_launch(void* const* d_in, const int* in_sizes, int n_in,
                              void* d_out, int out_size, void* d_ws, size_t ws_size,
                              hipStream_t stream) {
    const float* X = (const float*)d_in[0];
    // d_in[1] = indices: unused by the reference as well
    const int*   T = (const int*)d_in[2];
    const float* P = (const float*)d_in[3];
    float* out = (float*)d_out;

    // workspace: [accum f32 | pad to 256B | Pn bf16 512x512 | Xn bf16 N x 512]
    float*    accum = (float*)d_ws;
    unsigned* Pb    = (unsigned*)((char*)d_ws + 256);
    unsigned* Xb    = (unsigned*)((char*)d_ws + 256 + NB_CLASS * SZ_EMBED * 2);

    k_prep_proxies<<<NB_CLASS, 256, 0, stream>>>(P, Pb, accum);
    k_prep_x<<<NROWS / 8, 256, 0, stream>>>(X, Xb);
    const int tiles = (2 * NROWS) / TILE_M;            // 8192
    k_main<<<tiles / WPB, WPB * 32, 0, stream>>>(T, Xb, Pb, accum);
    k_finalize<<<1, 1, 0, stream>>>(accum, out);
    (void)in_sizes; (void)n_in; (void)out_size; (void)ws_size;
}
